// DisparityDecoder_51599737094157
// MI455X (gfx1250) — matrix operand order repaired and checked by reading, not compiled, not run on hardware
//
#include <hip/hip_runtime.h>
#include <cstddef>
#include <cstdint>

// ---------------------------------------------------------------------------
// MI455X (gfx1250) implementation of the stereo disparity decoder.
// - All GEMM-shaped work (1x1/3x3 convs as implicit GEMM, ConvNeXt MLP) runs
//   through v_wmma_f32_16x16x32_f16 (wave32, f16 A/B, f32 accumulate).
// - ConvNeXt LN->MLP->residual fused into one LDS-resident kernel (kills the
//   805MB 4x-expanded intermediate); its tile load uses the gfx1250 async
//   global->LDS path (global_load_async_to_lds_b128 + s_wait_asynccnt).
// - Depthwise 7x7 is LDS-tiled (22x22 halo per 16x16 output tile).
// ---------------------------------------------------------------------------

typedef _Float16 f16_t;
typedef __attribute__((ext_vector_type(8)))  f16_t v8h;
typedef __attribute__((ext_vector_type(16))) f16_t v16h;
typedef __attribute__((ext_vector_type(8)))  float v8f;

// Branchless exact-form GELU: erf via Abramowitz-Stegun 7.1.26 (|err|<1.5e-7),
// straight-line VALU (v_rcp + v_exp + FMAs) -- no EXEC divergence in hot loops.
__device__ __forceinline__ float gelu_f(float x) {
  const float ax = fabsf(x) * 0.70710678118654752440f;   // |x|/sqrt(2)
  const float t  = __builtin_amdgcn_rcpf(1.0f + 0.3275911f * ax);
  const float poly = t * (0.254829592f +
                     t * (-0.284496736f +
                     t * (1.421413741f +
                     t * (-1.453152027f +
                     t * 1.061405429f))));
  const float e = __builtin_amdgcn_exp2f(-ax * ax * 1.4426950408889634f);
  float erfv = copysignf(1.0f - poly * e, x);
  return 0.5f * x * (1.0f + erfv);
}

__device__ __forceinline__ v16h cat16(v8h lo, v8h hi) {
  return __builtin_shufflevector(lo, hi, 0,1,2,3,4,5,6,7,8,9,10,11,12,13,14,15);
}

__device__ __forceinline__ v8f wmma16(v16h a, v16h b, v8f c) {
  // (neg_a, A, neg_b, B, c_mod, C, reuse_a, reuse_b)
  return __builtin_amdgcn_wmma_f32_16x16x32_f16(false, a, false, b, (short)0, c,
                                                false, false);
}

// ------------------------------ f32 -> f16 weight conversion ----------------
__global__ __launch_bounds__(256) void k_cvt(const float* __restrict__ src,
                                             f16_t* __restrict__ dst, size_t n) {
  size_t i = (size_t)blockIdx.x * 256 + threadIdx.x;
  if (i < n) dst[i] = (f16_t)src[i];
}

// ------------------------------ L2 norm over channels (per pixel) -----------
__global__ __launch_bounds__(256) void k_l2norm(const float* __restrict__ in,
                                                float* __restrict__ out,
                                                int B, int C, int H, int W) {
  const int HW = H * W, Npx = B * HW;
  const int wv = threadIdx.x >> 5, lane = threadIdx.x & 31;
  const int p = blockIdx.x * 8 + wv;
  if (p >= Npx) return;
  const int b = p / HW, r = p % HW;
  const float* ib = in  + (size_t)b * C * HW + r;
  float*       ob = out + (size_t)b * C * HW + r;
  float s = 0.f;
  for (int c = lane; c < C; c += 32) { float v = ib[(size_t)c * HW]; s += v * v; }
  for (int o = 16; o > 0; o >>= 1) s += __shfl_xor(s, o, 32);
  const float inv = 1.0f / fmaxf(sqrtf(s), 1e-12f);
  for (int c = lane; c < C; c += 32) ob[(size_t)c * HW] = ib[(size_t)c * HW] * inv;
}

// ------------------------------ correlation volume --------------------------
__global__ __launch_bounds__(256) void k_corr(const float* __restrict__ nl,
                                              const float* __restrict__ nr,
                                              float* __restrict__ out,
                                              int B, int C, int H, int W,
                                              int Ctot, int cOff) {
  const int HW = H * W;
  const int Nout = B * 16 * HW;
  const int wv = threadIdx.x >> 5, lane = threadIdx.x & 31;
  const int idx = blockIdx.x * 8 + wv;
  if (idx >= Nout) return;
  int x = idx % W; int t = idx / W;
  int y = t % H;  t /= H;
  int d = t % 16; int b = t / 16;
  float s = 0.f;
  if (x >= d) {
    const float* lp = nl + (size_t)b * C * HW + y * W + x;
    const float* rp = nr + (size_t)b * C * HW + y * W + (x - d);
    for (int c = lane; c < C; c += 32)
      s += lp[(size_t)c * HW] * rp[(size_t)c * HW];
  }
  for (int o = 16; o > 0; o >>= 1) s += __shfl_xor(s, o, 32);
  if (lane == 0)
    out[(((size_t)b * Ctot + cOff + d) * H + y) * W + x] = s;
}

// ------------------------------ LayerNorm(+GELU) over channels, in place ----
__global__ __launch_bounds__(256) void k_ln_gelu(float* __restrict__ buf,
                                                 const float* __restrict__ g,
                                                 const float* __restrict__ bt,
                                                 int B, int C, int H, int W,
                                                 int doGelu) {
  const int HW = H * W, Npx = B * HW;
  const int wv = threadIdx.x >> 5, lane = threadIdx.x & 31;
  const int p = blockIdx.x * 8 + wv;
  if (p >= Npx) return;
  const int b = p / HW, r = p % HW;
  float* base = buf + (size_t)b * C * HW + r;
  float s = 0.f, s2 = 0.f;
  for (int c = lane; c < C; c += 32) {
    float v = base[(size_t)c * HW]; s += v; s2 += v * v;
  }
  for (int o = 16; o > 0; o >>= 1) { s += __shfl_xor(s, o, 32); s2 += __shfl_xor(s2, o, 32); }
  const float mu  = s / (float)C;
  const float var = s2 / (float)C - mu * mu;
  const float inv = rsqrtf(var + 1e-6f);
  for (int c = lane; c < C; c += 32) {
    float v = (base[(size_t)c * HW] - mu) * inv * g[c] + bt[c];
    if (doGelu) v = gelu_f(v);
    base[(size_t)c * HW] = v;
  }
}

// ------------------------------ 2x bilinear upsample (linspace variant) -----
__global__ __launch_bounds__(256) void k_up2(const float* __restrict__ in,
                                             float* __restrict__ out,
                                             int B, int C, int H, int W,
                                             int Ctot, int cOff) {
  const int Ho = 2 * H, Wo = 2 * W;
  const size_t total = (size_t)B * C * Ho * Wo;
  size_t n = (size_t)blockIdx.x * 256 + threadIdx.x;
  if (n >= total) return;
  int ox = (int)(n % Wo); size_t t = n / Wo;
  int oy = (int)(t % Ho); t /= Ho;
  int c  = (int)(t % C);  int b = (int)(t / C);
  const float fy = (float)oy * (float)(H - 1) / (float)(Ho - 1);
  const float fx = (float)ox * (float)(W - 1) / (float)(Wo - 1);
  int y0 = (int)floorf(fy); int y1 = min(y0 + 1, H - 1); float wy = fy - (float)y0;
  int x0 = (int)floorf(fx); int x1 = min(x0 + 1, W - 1); float wx = fx - (float)x0;
  const float* ib = in + ((size_t)b * C + c) * H * W;
  float r0 = ib[(size_t)y0 * W + x0] * (1.f - wx) + ib[(size_t)y0 * W + x1] * wx;
  float r1 = ib[(size_t)y1 * W + x0] * (1.f - wx) + ib[(size_t)y1 * W + x1] * wx;
  out[(((size_t)b * Ctot + cOff + c) * Ho + oy) * Wo + ox] = r0 * (1.f - wy) + r1 * wy;
}

// ------------------------------ copy channels into a concat buffer ----------
__global__ __launch_bounds__(256) void k_copy_ch(const float* __restrict__ src,
                                                 float* __restrict__ dst,
                                                 int B, int Cs, int H, int W,
                                                 int Ctot, int cOff) {
  const size_t total = (size_t)B * Cs * H * W;
  size_t i = (size_t)blockIdx.x * 256 + threadIdx.x;
  if (i >= total) return;
  int x = (int)(i % W); size_t t = i / W;
  int y = (int)(t % H); t /= H;
  int c = (int)(t % Cs); int b = (int)(t / Cs);
  dst[(((size_t)b * Ctot + cOff + c) * H + y) * W + x] = src[i];
}

// ------------------------------ depthwise 7x7, pad 3, LDS-tiled -------------
// 16x16 output tile per block, 22x22 input halo staged in LDS, filter in LDS.
// Cuts global load traffic ~26x vs the naive per-output gather.
__global__ __launch_bounds__(256) void k_dw7_tiled(const float* __restrict__ in,
                                                   const float* __restrict__ w,
                                                   float* __restrict__ out,
                                                   int B, int C, int H, int W) {
  __shared__ float tile[22 * 22];
  __shared__ float wc_s[49];
  const int tx = threadIdx.x & 15, ty = threadIdx.x >> 4;
  const int tilesX = W >> 4, tilesY = H >> 4;
  int bid = blockIdx.x;
  const int tX = bid % tilesX; bid /= tilesX;
  const int tY = bid % tilesY; bid /= tilesY;
  const int c = bid % C; const int b = bid / C;
  const float* ib = in + ((size_t)b * C + c) * H * W;
  if (threadIdx.x < 49) wc_s[threadIdx.x] = w[(size_t)c * 49 + threadIdx.x];
  const int y0 = tY * 16 - 3, x0 = tX * 16 - 3;
  for (int idx = threadIdx.x; idx < 22 * 22; idx += 256) {
    int lx = idx % 22, ly = idx / 22;
    int gy = y0 + ly, gx = x0 + lx;
    float v = 0.f;
    if (gy >= 0 && gy < H && gx >= 0 && gx < W) v = ib[(size_t)gy * W + gx];
    tile[idx] = v;
  }
  __syncthreads();
  float s = 0.f;
  #pragma unroll
  for (int ky = 0; ky < 7; ++ky) {
    #pragma unroll
    for (int kx = 0; kx < 7; ++kx)
      s += tile[(ty + ky) * 22 + (tx + kx)] * wc_s[ky * 7 + kx];
  }
  out[((size_t)b * C + c) * H * W + (size_t)(tY * 16 + ty) * W + (tX * 16 + tx)] = s;
}

// ------------------------------ implicit-GEMM conv via WMMA -----------------
// D(16 cout x 16 pixels) per wave. M=Cout (mult of 16), N=B*H*W (mult of 16),
// K=Cin*KH*KW (mult of 32). Weights pre-converted to f16, row-major [Cout][K].
template <int KH, int KW>
__global__ __launch_bounds__(128) void k_conv_wmma(
    const float* __restrict__ in, const f16_t* __restrict__ w16,
    float* __restrict__ out,
    int B, int Cin, int H, int W, int Cout, int pad, int Ctot, int cOff) {
  const int lane = threadIdx.x;       // 0..31
  const int wv   = threadIdx.y;       // 0..3
  const int HW   = H * W;
  const int Npx  = B * HW;
  const int pixTile = blockIdx.x * 4 + wv;
  if (pixTile * 16 >= Npx) return;    // wave-uniform
  const int coutBase = blockIdx.y * 16;
  const int K = Cin * KH * KW;
  const int laneHalf = lane >> 4;     // 0/1
  const int lm = lane & 15;
  const int p = pixTile * 16 + lm;    // this lane's output pixel (GEMM column)
  const int b = p / HW, rp = p % HW;
  const int py = rp / W, px = rp % W;
  const float* inB = in + (size_t)b * Cin * HW;
  // A row: weights for cout = coutBase+lm; k = kb + laneHalf*8 + {0..7,16..23}
  const f16_t* wrow = w16 + (size_t)(coutBase + lm) * K + laneHalf * 8;
  v8f acc = {0.f, 0.f, 0.f, 0.f, 0.f, 0.f, 0.f, 0.f};
  for (int kb = 0; kb < K; kb += 32) {
    __builtin_prefetch(wrow + kb + 512, 0, 0);   // global_prefetch_b8, next rows
    v8h a0 = *(const v8h*)(wrow + kb);
    v8h a1 = *(const v8h*)(wrow + kb + 16);
    v16h afrag = cat16(a0, a1);
    // B column: im2col gather; k = kb + laneHalf*8 + {0..7,16..23}
    v16h bfrag;
    #pragma unroll
    for (int i = 0; i < 16; ++i) {
      const int k = kb + laneHalf * 8 + (i & 7) + ((i >> 3) << 4);
      const int cin = k / (KH * KW);
      const int rr  = k % (KH * KW);
      const int ky = rr / KW, kx = rr % KW;
      const int iy = py + ky - pad, ix = px + kx - pad;
      float v = 0.f;
      if (iy >= 0 && iy < H && ix >= 0 && ix < W)
        v = inB[(size_t)cin * HW + (size_t)iy * W + ix];
      bfrag[i] = (f16_t)v;
    }
    acc = wmma16(afrag, bfrag, acc);
  }
  // D: lane holds N=lm; VGPR v -> M = v + laneHalf*8
  float* ob = out + ((size_t)b * Ctot + cOff) * HW + (size_t)py * W + px;
  #pragma unroll
  for (int v = 0; v < 8; ++v) {
    const int m = coutBase + v + laneHalf * 8;
    ob[(size_t)m * HW] = acc[v];
  }
}

// ------------------------------ fused ConvNeXt LN -> MLP -> residual --------
// Per 16-pixel tile: async global->LDS tile load, LN over 96 ch, GEMM1
// (384x96)@(96x16)+b1, GELU -> LDS f16, GEMM2 (96x384)@(384x16)+b2, residual.
#define NXC 96
#define NXD 384
#define BSTR 104   // LDS f16 stride for K=96 stage (conflict-free, 16B aligned)
#define MSTR 392   // LDS f16 stride for K=384 stage
__global__ __launch_bounds__(256) void k_nx_fused(
    const float* __restrict__ h,      // dwconv output (B,96,H,W)
    const float* __restrict__ g, const float* __restrict__ bt,  // LN gamma/beta
    const f16_t* __restrict__ w1, const float* __restrict__ b1, // (384,96)
    const f16_t* __restrict__ w2, const float* __restrict__ b2, // (96,384)
    float* __restrict__ xio,          // residual in / out (in place)
    int B, int H, int W) {
  __shared__ float  raw[16 * NXC];        // [c][n] : 96 rows of 16 pixels
  __shared__ float  mu_s[16], inv_s[16];
  __shared__ f16_t  nb[16 * BSTR];        // normalized, [n][k]
  __shared__ f16_t  mid[16 * MSTR];       // gelu(W1 h + b1), [n][k]
  const int tid = threadIdx.x;
  const int HW = H * W;
  const int pix0 = blockIdx.x * 16;
  const int b0 = pix0 / HW, r0 = pix0 % HW;   // 16 pixels contiguous, same batch
  // --- phase 1: async copy 96 x 64B rows straight into LDS (ASYNCcnt path).
  // Each lane moves 16B; generic->LDS address is the low 32 bits (ISA 10.2).
  for (int idx = tid; idx < NXC * 4; idx += 256) {
    const int c = idx >> 2, q = idx & 3;
    const float* gp = h + ((size_t)b0 * NXC + c) * HW + r0 + q * 4;
    unsigned lds_addr = (unsigned)(uintptr_t)(raw + c * 16 + q * 4);
    unsigned long long gaddr = (unsigned long long)(uintptr_t)gp;
    asm volatile("global_load_async_to_lds_b128 %0, %1, off"
                 :: "v"(lds_addr), "v"(gaddr) : "memory");
  }
  asm volatile("s_wait_asynccnt 0x0" ::: "memory");
  __syncthreads();
  // --- per-pixel LN statistics
  if (tid < 16) {
    float s = 0.f, s2 = 0.f;
    for (int c = 0; c < NXC; ++c) { float v = raw[(c << 4) + tid]; s += v; s2 += v * v; }
    const float mu = s / (float)NXC;
    mu_s[tid] = mu;
    inv_s[tid] = rsqrtf(s2 / (float)NXC - mu * mu + 1e-6f);
  }
  __syncthreads();
  for (int idx = tid; idx < 16 * NXC; idx += 256) {
    const int n = idx & 15, c = idx >> 4;
    const float v = (raw[(c << 4) + n] - mu_s[n]) * inv_s[n] * g[c] + bt[c];
    nb[n * BSTR + c] = (f16_t)v;
  }
  __syncthreads();
  const int lane = tid & 31, wv = tid >> 5;
  const int laneHalf = lane >> 4, lm = lane & 15;
  // --- GEMM1: 24 M-tiles over 8 waves, K=96
  for (int mt = wv; mt < NXD / 16; mt += 8) {
    const f16_t* wrow = w1 + (size_t)(mt * 16 + lm) * NXC + laneHalf * 8;
    const f16_t* brow = nb + lm * BSTR + laneHalf * 8;
    v8f acc = {0.f, 0.f, 0.f, 0.f, 0.f, 0.f, 0.f, 0.f};
    #pragma unroll
    for (int kb = 0; kb < NXC; kb += 32) {
      v16h afrag = cat16(*(const v8h*)(wrow + kb), *(const v8h*)(wrow + kb + 16));
      v16h bfrag = cat16(*(const v8h*)(brow + kb), *(const v8h*)(brow + kb + 16));
      acc = wmma16(afrag, bfrag, acc);
    }
    #pragma unroll
    for (int v = 0; v < 8; ++v) {
      const int m = mt * 16 + v + laneHalf * 8;
      mid[lm * MSTR + m] = (f16_t)gelu_f(acc[v] + b1[m]);
    }
  }
  __syncthreads();
  // --- GEMM2: 6 M-tiles on waves 0..5, K=384; fused +b2 and residual add
  if (wv < 6) {
    const int mt = wv;
    const f16_t* wrow = w2 + (size_t)(mt * 16 + lm) * NXD + laneHalf * 8;
    const f16_t* brow = mid + lm * MSTR + laneHalf * 8;
    v8f acc = {0.f, 0.f, 0.f, 0.f, 0.f, 0.f, 0.f, 0.f};
    for (int kb = 0; kb < NXD; kb += 32) {
      v16h afrag = cat16(*(const v8h*)(wrow + kb), *(const v8h*)(wrow + kb + 16));
      v16h bfrag = cat16(*(const v8h*)(brow + kb), *(const v8h*)(brow + kb + 16));
      acc = wmma16(afrag, bfrag, acc);
    }
    const int p = pix0 + lm;
    const int b = p / HW, r = p % HW;
    float* ob = xio + (size_t)b * NXC * HW + r;
    #pragma unroll
    for (int v = 0; v < 8; ++v) {
      const int m = mt * 16 + v + laneHalf * 8;
      ob[(size_t)m * HW] += acc[v] + b2[m];   // in-place residual
    }
  }
}

// ===========================================================================
extern "C" void kernel_launch(void* const* d_in, const int* in_sizes, int n_in,
                              void* d_out, int out_size, void* d_ws, size_t ws_size,
                              hipStream_t stream) {
  (void)in_sizes; (void)n_in; (void)out_size;
  // ---- inputs ----
  const float* left0 = (const float*)d_in[0];   // (4, 96, 64,128)
  const float* left1 = (const float*)d_in[1];   // (4,192, 32, 64)
  const float* left2 = (const float*)d_in[2];   // (4,384, 16, 32)
  const float* left3 = (const float*)d_in[3];   // (4,768,  8, 16)
  const float* right3 = (const float*)d_in[7];  // (4,768,  8, 16)
  // ---- params: jax pytree leaves, dict keys sorted recursively ----
  const float* ad_b  = (const float*)d_in[8];
  const float* ad_g  = (const float*)d_in[9];
  const float* ad_w  = (const float*)d_in[10];  // (768,32,3,3)
  const float* b0_b1 = (const float*)d_in[11];
  const float* b0_b2 = (const float*)d_in[12];
  const float* b0_g1 = (const float*)d_in[13];
  const float* b0_g2 = (const float*)d_in[14];
  const float* b0_w1 = (const float*)d_in[15];  // (384,1152,3,3)
  const float* b0_w2 = (const float*)d_in[16];  // (384, 384,3,3)
  const float* b1_b1 = (const float*)d_in[17];
  const float* b1_b2 = (const float*)d_in[18];
  const float* b1_g1 = (const float*)d_in[19];
  const float* b1_g2 = (const float*)d_in[20];
  const float* b1_w1 = (const float*)d_in[21];  // (192,576,3,3)
  const float* b1_w2 = (const float*)d_in[22];  // (192,192,3,3)
  const float* b2_b1 = (const float*)d_in[23];
  const float* b2_b2 = (const float*)d_in[24];
  const float* b2_g1 = (const float*)d_in[25];
  const float* b2_g2 = (const float*)d_in[26];
  const float* b2_w1 = (const float*)d_in[27];  // (96,288,3,3)
  const float* b2_w2 = (const float*)d_in[28];  // (96, 96,3,3)
  const float* n0_b  = (const float*)d_in[29];
  const float* n0_b1 = (const float*)d_in[30];
  const float* n0_b2 = (const float*)d_in[31];
  const float* n0_dw = (const float*)d_in[32];  // (96,1,7,7)
  const float* n0_g  = (const float*)d_in[33];
  const float* n0_w1 = (const float*)d_in[34];  // (384,96)
  const float* n0_w2 = (const float*)d_in[35];  // (96,384)
  const float* n1_b  = (const float*)d_in[36];
  const float* n1_b1 = (const float*)d_in[37];
  const float* n1_b2 = (const float*)d_in[38];
  const float* n1_dw = (const float*)d_in[39];
  const float* n1_g  = (const float*)d_in[40];
  const float* n1_w1 = (const float*)d_in[41];
  const float* n1_w2 = (const float*)d_in[42];
  const float* sq_b  = (const float*)d_in[43];
  const float* sq_g  = (const float*)d_in[44];
  const float* sq_w  = (const float*)d_in[45];  // (16,768,1,1)

  // ---- workspace layout ----
  char* ws = (char*)d_ws;
  size_t ho = 0;
  auto allocH = [&](size_t n) { f16_t* p = (f16_t*)(ws + ho); ho += n * sizeof(f16_t); return p; };
  f16_t* sqWh   = allocH(16 * 768);
  f16_t* adWh   = allocH(768 * 288);
  f16_t* b0w1h  = allocH(384 * 10368);
  f16_t* b0w2h  = allocH(384 * 3456);
  f16_t* b1w1h  = allocH(192 * 5184);
  f16_t* b1w2h  = allocH(192 * 1728);
  f16_t* b2w1h  = allocH(96 * 2592);
  f16_t* b2w2h  = allocH(96 * 864);
  f16_t* n0w1h  = allocH(384 * 96);
  f16_t* n0w2h  = allocH(96 * 384);
  f16_t* n1w1h  = allocH(384 * 96);
  f16_t* n1w2h  = allocH(96 * 384);
  size_t fo = (ho + 255) & ~(size_t)255;
  auto allocF = [&](size_t n) { float* p = (float*)(ws + fo); fo += n * sizeof(float); return p; };
  const size_t BIG = 50331648;         // 4*96*256*512
  float* INTER = allocF(3145728);      // intercept (4,96,64,128)
  float* X = allocF(BIG);
  float* Y = allocF(BIG);
  if (ws_size < fo) return;            // insufficient scratch; bail deterministically
  // small level-3 temporaries live inside Y (dead before Y's first big use)
  float* nl  = Y;              // 4*768*8*16 = 393216
  float* nr  = Y + 393216;     // 393216
  float* x32 = Y + 786432;     // (4,32,8,16) = 16384
  float* sqt = Y + 802816;     // (4,16,8,16) = 8192

  // ---- helpers ----
  auto cvt = [&](const float* s, f16_t* d, size_t n) {
    k_cvt<<<(unsigned)((n + 255) / 256), 256, 0, stream>>>(s, d, n);
  };
  auto conv3 = [&](const float* in, const f16_t* w, float* out,
                   int Cin, int Hh, int Ww, int Cout, int Ctot, int cOff) {
    int Npx = 4 * Hh * Ww;
    dim3 grid((Npx / 16 + 3) / 4, Cout / 16);
    k_conv_wmma<3, 3><<<grid, dim3(32, 4), 0, stream>>>(in, w, out, 4, Cin, Hh, Ww,
                                                        Cout, 1, Ctot, cOff);
  };
  auto conv1 = [&](const float* in, const f16_t* w, float* out,
                   int Cin, int Hh, int Ww, int Cout, int Ctot, int cOff) {
    int Npx = 4 * Hh * Ww;
    dim3 grid((Npx / 16 + 3) / 4, Cout / 16);
    k_conv_wmma<1, 1><<<grid, dim3(32, 4), 0, stream>>>(in, w, out, 4, Cin, Hh, Ww,
                                                        Cout, 0, Ctot, cOff);
  };
  auto lng = [&](float* buf, const float* gg, const float* bb, int C, int Hh, int Ww) {
    int Npx = 4 * Hh * Ww;
    k_ln_gelu<<<(Npx + 7) / 8, 256, 0, stream>>>(buf, gg, bb, 4, C, Hh, Ww, 1);
  };
  auto up2 = [&](const float* in, float* out, int C, int Hh, int Ww, int Ctot) {
    size_t total = (size_t)4 * C * 2 * Hh * 2 * Ww;
    k_up2<<<(unsigned)((total + 255) / 256), 256, 0, stream>>>(in, out, 4, C, Hh, Ww, Ctot, 0);
  };
  auto copych = [&](const float* s, float* d, int Cs, int Hh, int Ww, int Ctot, int cOff) {
    size_t total = (size_t)4 * Cs * Hh * Ww;
    k_copy_ch<<<(unsigned)((total + 255) / 256), 256, 0, stream>>>(s, d, 4, Cs, Hh, Ww, Ctot, cOff);
  };
  auto dw7 = [&](const float* in, const float* w, float* out, int Hh, int Ww) {
    unsigned blocks = (unsigned)(4 * 96 * (Hh >> 4) * (Ww >> 4));
    k_dw7_tiled<<<blocks, 256, 0, stream>>>(in, w, out, 4, 96, Hh, Ww);
  };

  // ---- 0) weights -> f16 (A-matrices for WMMA), once per launch ----
  cvt(sq_w, sqWh, 16 * 768);       cvt(ad_w, adWh, 768 * 288);
  cvt(b0_w1, b0w1h, 384 * 10368);  cvt(b0_w2, b0w2h, 384 * 3456);
  cvt(b1_w1, b1w1h, 192 * 5184);   cvt(b1_w2, b1w2h, 192 * 1728);
  cvt(b2_w1, b2w1h, 96 * 2592);    cvt(b2_w2, b2w2h, 96 * 864);
  cvt(n0_w1, n0w1h, 384 * 96);     cvt(n0_w2, n0w2h, 96 * 384);
  cvt(n1_w1, n1w1h, 384 * 96);     cvt(n1_w2, n1w2h, 96 * 384);

  // ---- 1) level-3 head: correlation volume + squeeze, concat -> adapter ----
  k_l2norm<<<(512 + 7) / 8, 256, 0, stream>>>(left3, nl, 4, 768, 8, 16);
  k_l2norm<<<(512 + 7) / 8, 256, 0, stream>>>(right3, nr, 4, 768, 8, 16);
  k_corr<<<(8192 + 7) / 8, 256, 0, stream>>>(nl, nr, x32, 4, 768, 8, 16, 32, 16);
  conv1(left3, sqWh, sqt, 768, 8, 16, 16, 16, 0);
  lng(sqt, sq_g, sq_b, 16, 8, 16);
  copych(sqt, x32, 16, 8, 16, 32, 0);
  conv3(x32, adWh, X, 32, 8, 16, 768, 768, 0);     // adapter 32->768
  lng(X, ad_g, ad_b, 768, 8, 16);                  // x in X (4,768,8,16)

  // ---- 2) decoder block 0: 8x16 -> 16x32 ----
  up2(X, Y, 768, 8, 16, 1152);
  copych(left2, Y, 384, 16, 32, 1152, 768);
  conv3(Y, b0w1h, X, 1152, 16, 32, 384, 384, 0);  lng(X, b0_g1, b0_b1, 384, 16, 32);
  conv3(X, b0w2h, Y, 384, 16, 32, 384, 384, 0);   lng(Y, b0_g2, b0_b2, 384, 16, 32);

  // ---- 3) decoder block 1: 16x32 -> 32x64 ----
  up2(Y, X, 384, 16, 32, 576);
  copych(left1, X, 192, 32, 64, 576, 384);
  conv3(X, b1w1h, Y, 576, 32, 64, 192, 192, 0);   lng(Y, b1_g1, b1_b1, 192, 32, 64);
  conv3(Y, b1w2h, X, 192, 32, 64, 192, 192, 0);   lng(X, b1_g2, b1_b2, 192, 32, 64);

  // ---- 4) decoder block 2: 32x64 -> 64x128 (-> intercept) ----
  up2(X, Y, 192, 32, 64, 288);
  copych(left0, Y, 96, 64, 128, 288, 192);
  conv3(Y, b2w1h, X, 288, 64, 128, 96, 96, 0);    lng(X, b2_g1, b2_b1, 96, 64, 128);
  conv3(X, b2w2h, INTER, 96, 64, 128, 96, 96, 0); lng(INTER, b2_g2, b2_b2, 96, 64, 128);

  // ---- 5) ConvNeXt stage 0: 64x128 -> 128x256 ----
  up2(INTER, X, 96, 64, 128, 96);                  // X = (4,96,128,256)
  dw7(X, n0_dw, Y, 128, 256);
  k_nx_fused<<<4 * 128 * 256 / 16, 256, 0, stream>>>(Y, n0_g, n0_b, n0w1h, n0_b1,
                                                     n0w2h, n0_b2, X, 4, 128, 256);

  // ---- 6) ConvNeXt stage 1: 128x256 -> 256x512 ----
  up2(X, Y, 96, 128, 256, 96);                     // Y = (4,96,256,512)
  dw7(Y, n1_dw, X, 256, 512);
  k_nx_fused<<<4 * 256 * 512 / 16, 256, 0, stream>>>(X, n1_g, n1_b, n1w1h, n1_b1,
                                                     n1w2h, n1_b2, Y, 4, 256, 512);

  // ---- 7) assemble outputs: (final x, intercept) concatenated ----
  hipMemcpyAsync(d_out, Y, BIG * sizeof(float), hipMemcpyDeviceToDevice, stream);
  hipMemcpyAsync((float*)d_out + BIG, INTER, (size_t)3145728 * sizeof(float),
                 hipMemcpyDeviceToDevice, stream);
}
